// NodeEdgeCoord_AE_55508157334089
// MI455X (gfx1250) — matrix operand order repaired and checked by reading, not compiled, not run on hardware
//
#include <hip/hip_runtime.h>
#include <hip/hip_bf16.h>

#define NNODE 10000
#define NEDGE 320000

typedef __attribute__((ext_vector_type(16))) __bf16 v16bf;
typedef __attribute__((ext_vector_type(8)))  float  v8f;
typedef __attribute__((ext_vector_type(4)))  int    v4i;

#define AS1 __attribute__((address_space(1)))
#define AS3 __attribute__((address_space(3)))

#if __has_builtin(__builtin_amdgcn_global_load_async_to_lds_b128)
#define HAVE_ASYNC_LDS 1
#endif

// Native f32 -> bf16 (backend emits v_cvt_pk_bf16_f32 for pairs)
__device__ __forceinline__ __bf16 f2bf(float f) { return (__bf16)f; }

union FragU { uint4 q[2]; v16bf v; };

// A-fragment: 16x32 bf16 tile from row-major LDS [stride in elements].
// lane(0..15)=rows M, g=lane>>4 picks K {g*8..g*8+7, 16+g*8..16+g*8+7}
__device__ __forceinline__ v16bf lds_frag_A(const __bf16* base, int stride) {
  const int lane = threadIdx.x & 31;
  const __bf16* p = base + (lane & 15) * stride + (lane >> 4) * 8;
  FragU f;
  f.q[0] = *(const uint4*)(p);
  f.q[1] = *(const uint4*)(p + 16);
  return f.v;
}

// B-fragment from transposed weights Wt[n][k] (row-major over n).
// lane(0..15)=cols N, g=lane>>4 picks K {g*8..g*8+7, 16+g*8..16+g*8+7}
__device__ __forceinline__ v16bf lds_frag_B(const __bf16* baseT, int stride) {
  const int lane = threadIdx.x & 31;
  const __bf16* p = baseT + (lane & 15) * stride + (lane >> 4) * 8;
  FragU f;
  f.q[0] = *(const uint4*)(p);
  f.q[1] = *(const uint4*)(p + 16);
  return f.v;
}

__device__ __forceinline__ v8f wmma_bf16(v16bf a, v16bf b, v8f c) {
  return __builtin_amdgcn_wmma_f32_16x16x32_bf16(false, a, false, b, (short)0, c, false, false);
}

// Stage pre-converted bf16 weights from global into LDS.
// Async (gfx1250 ASYNCcnt path) when available; coalesced b128 otherwise.
__device__ __forceinline__ void copy_w(__bf16* dst, const __bf16* src, int nelem, int t) {
#ifdef HAVE_ASYNC_LDS
  for (int idx = t * 8; idx < nelem; idx += 256 * 8)
    __builtin_amdgcn_global_load_async_to_lds_b128(
        (AS1 v4i*)(src + idx), (AS3 v4i*)(dst + idx), 0, 0);
#else
  for (int idx = t; idx < nelem / 8; idx += 256)
    ((uint4*)dst)[idx] = ((const uint4*)src)[idx];
#endif
}

__device__ __forceinline__ void wait_async_copies() {
#ifdef HAVE_ASYNC_LDS
#if __has_builtin(__builtin_amdgcn_s_wait_asynccnt)
  __builtin_amdgcn_s_wait_asynccnt(0);
#else
  asm volatile("s_wait_asynccnt 0" ::: "memory");
#endif
#endif
}

__global__ void zero_kernel(float* __restrict__ p, int n) {
  int i = blockIdx.x * blockDim.x + threadIdx.x;
  if (i < n) p[i] = 0.f;
}

// ---------------------------------------------------------------------------
// One-shot weight prep: transpose + convert to bf16 images in workspace.
// Layout (bf16 elems): [0)=ew1t 128x192, [24576)=ew2t 32x128,
//                      [28672)=nw1t 128x128, [45056)=nw2t 64x128  (total 53248)
// ---------------------------------------------------------------------------
__global__ void prep_weights(const float* __restrict__ ew1, const float* __restrict__ ew2,
                             const float* __restrict__ nw1, const float* __restrict__ nw2,
                             __bf16* __restrict__ out) {
  int i = blockIdx.x * 256 + threadIdx.x;
  if (i < 24576) {
    int n = i / 192, k = i - n * 192;
    out[i] = f2bf(k < 161 ? ew1[k * 128 + n] : 0.f);
  } else if (i < 28672) {
    int j = i - 24576; int n = j >> 7, k = j & 127;
    out[i] = f2bf(ew2[k * 32 + n]);
  } else if (i < 45056) {
    int j = i - 28672; int n = j >> 7, k = j & 127;
    out[i] = f2bf(k < 99 ? nw1[k * 128 + n] : 0.f);
  } else if (i < 53248) {
    int j = i - 45056; int n = j >> 7, k = j & 127;
    out[i] = f2bf(nw2[k * 64 + n]);
  }
}

// ---------------------------------------------------------------------------
// Edge kernel: gather -> edge MLP (WMMA bf16) -> coord MLP + atomics + heads
// 128 edges per workgroup, 256 threads (8 waves, one 16-row M-tile per wave)
// ---------------------------------------------------------------------------
struct EdgePost { float EO[128 * 32]; float Emb[128 * 8]; };
union EdgeScratch { __bf16 A[128 * 192]; EdgePost post; };

__global__ __launch_bounds__(256) void edge_kernel(
    const float* __restrict__ node_feats, const int* __restrict__ eidx,
    const float* __restrict__ edge_attr, const float* __restrict__ coords,
    const __bf16* __restrict__ w1t, const float* __restrict__ b1,
    const __bf16* __restrict__ w2t, const float* __restrict__ b2,
    const float* __restrict__ cw1, const float* __restrict__ cb1,
    const float* __restrict__ cw2,
    const float* __restrict__ fcw, const float* __restrict__ fcb,
    const float* __restrict__ dw,  const float* __restrict__ db,
    float* __restrict__ edge_out_ws, float* __restrict__ agg,
    float* __restrict__ coordacc,
    float* __restrict__ edge_emb_o, float* __restrict__ recon_edge_o)
{
  __shared__ EdgeScratch u;                  // 48 KB (A) / 20 KB (post)
  __shared__ __bf16 sW1[128 * 192];          // Wt[n][k], 48 KB
  __shared__ __bf16 sH [128 * 128];          // hidden, 32 KB
  __shared__ __bf16 sW2[32 * 128];           // Wt[n][k], 8 KB
  __shared__ float sB1[128], sB2[32], sCW1[64], sCB1[2], sCW2[2];
  __shared__ float sFCW[256], sFCB[8], sDW[256], sDB[32];

  const int t  = threadIdx.x;
  const int e0 = blockIdx.x * 128;

  // --- stage pre-transposed bf16 weights (async overlaps the gather) ---
  copy_w(sW1, w1t, 128 * 192, t);
  copy_w(sW2, w2t, 32 * 128, t);
  if (t < 128) sB1[t] = b1[t];
  if (t < 64)  sCW1[t] = cw1[t];
  if (t < 32)  { sB2[t] = b2[t]; sDB[t] = db[t]; }
  if (t < 8)   sFCB[t] = fcb[t];
  if (t < 2)   { sCB1[t] = cb1[t]; sCW2[t] = cw2[t]; }
  for (int idx = t; idx < 256; idx += 256) { sFCW[idx] = fcw[idx]; sDW[idx] = dw[idx]; }

  // --- gather edge_in rows into LDS (bf16), 2 threads per edge ---
  const int el = t >> 1, hf = t & 1;
  const int e  = e0 + el;
  const int er = eidx[e], ec = eidx[NEDGE + e];
  float nx = 0.f, ny = 0.f, nz = 0.f;
  __bf16* Ae = u.A + el * 192;
  if (hf == 0) {
    float dx = coords[er * 3 + 0] - coords[ec * 3 + 0];
    float dy = coords[er * 3 + 1] - coords[ec * 3 + 1];
    float dz = coords[er * 3 + 2] - coords[ec * 3 + 2];
    float rad = dx * dx + dy * dy + dz * dz;
    float inv = 1.f / (sqrtf(rad) + 1.f);
    nx = dx * inv; ny = dy * inv; nz = dz * inv;
    #pragma unroll 8
    for (int k = 0; k < 64; ++k) Ae[k] = f2bf(node_feats[er * 64 + k]);
    Ae[160] = f2bf(rad);
    #pragma unroll
    for (int k = 161; k < 192; ++k) Ae[k] = f2bf(0.f);
  } else {
    #pragma unroll 8
    for (int k = 0; k < 64; ++k) Ae[64 + k] = f2bf(node_feats[ec * 64 + k]);
    #pragma unroll 8
    for (int k = 0; k < 32; ++k) Ae[128 + k] = f2bf(edge_attr[e * 32 + k]);
  }
  wait_async_copies();
  __syncthreads();

  const int wave = t >> 5, lane = t & 31;
  const int ln = lane & 15, gh = lane >> 4;
  const int m0 = wave * 16;

  // --- layer 1: [16x192] x [192x128] -> ReLU -> sH ---
  for (int nt = 0; nt < 8; ++nt) {
    v8f acc; float bb = sB1[nt * 16 + ln];
    #pragma unroll
    for (int r = 0; r < 8; ++r) acc[r] = bb;
    #pragma unroll
    for (int kc = 0; kc < 6; ++kc) {
      v16bf a = lds_frag_A(u.A + m0 * 192 + kc * 32, 192);
      v16bf b = lds_frag_B(sW1 + (nt * 16) * 192 + kc * 32, 192);
      acc = wmma_bf16(a, b, acc);
    }
    #pragma unroll
    for (int r = 0; r < 8; ++r)
      sH[(m0 + gh * 8 + r) * 128 + nt * 16 + ln] = f2bf(fmaxf(acc[r], 0.f));
  }
  __syncthreads();

  // --- layer 2: [16x128] x [128x32] -> edge_out (LDS + global) ---
  for (int nt = 0; nt < 2; ++nt) {
    v8f acc; float bb = sB2[nt * 16 + ln];
    #pragma unroll
    for (int r = 0; r < 8; ++r) acc[r] = bb;
    #pragma unroll
    for (int kc = 0; kc < 4; ++kc) {
      v16bf a = lds_frag_A(sH + m0 * 128 + kc * 32, 128);
      v16bf b = lds_frag_B(sW2 + (nt * 16) * 128 + kc * 32, 128);
      acc = wmma_bf16(a, b, acc);
    }
    #pragma unroll
    for (int r = 0; r < 8; ++r) {
      int mm = m0 + gh * 8 + r, cc = nt * 16 + ln;
      u.post.EO[mm * 32 + cc] = acc[r];
      edge_out_ws[(size_t)(e0 + mm) * 32 + cc] = acc[r];
    }
  }
  __syncthreads();

  // --- epilogue: coord MLP, scatter atomics, edge heads ---
  const float* eo = u.post.EO + el * 32;
  if (hf == 0) {
    float h0 = sCB1[0], h1 = sCB1[1];
    #pragma unroll 8
    for (int k = 0; k < 32; ++k) { float v = eo[k]; h0 += v * sCW1[2 * k]; h1 += v * sCW1[2 * k + 1]; }
    float wv = fmaxf(h0, 0.f) * sCW2[0] + fmaxf(h1, 0.f) * sCW2[1];
    unsafeAtomicAdd(&coordacc[er * 3 + 0], nx * wv);
    unsafeAtomicAdd(&coordacc[er * 3 + 1], ny * wv);
    unsafeAtomicAdd(&coordacc[er * 3 + 2], nz * wv);
    #pragma unroll
    for (int j = 0; j < 8; ++j) {
      float s = sFCB[j];
      #pragma unroll 8
      for (int k = 0; k < 32; ++k) s += eo[k] * sFCW[k * 8 + j];
      u.post.Emb[el * 8 + j] = s;
      edge_emb_o[(size_t)e * 8 + j] = s;
    }
  }
  #pragma unroll
  for (int k = hf * 16; k < hf * 16 + 16; ++k)
    unsafeAtomicAdd(&agg[er * 32 + k], eo[k]);
  __syncthreads();
  const float* em = u.post.Emb + el * 8;
  #pragma unroll
  for (int j = hf * 16; j < hf * 16 + 16; ++j) {
    float s = sDB[j];
    #pragma unroll
    for (int k = 0; k < 8; ++k) s += em[k] * sDW[k * 32 + j];
    recon_edge_o[(size_t)e * 32 + j] = s;
  }
}

// ---------------------------------------------------------------------------
// Node kernel: concat -> node MLP (WMMA bf16) -> node_emb + recon + coord_out
// ---------------------------------------------------------------------------
union NodeScratch { __bf16 A[128 * 128]; float NO[128 * 64]; };

__global__ __launch_bounds__(256) void node_kernel(
    const float* __restrict__ node_feats, const float* __restrict__ coords,
    const float* __restrict__ agg, const float* __restrict__ coordacc,
    const __bf16* __restrict__ w1t, const float* __restrict__ b1,
    const __bf16* __restrict__ w2t, const float* __restrict__ b2,
    const float* __restrict__ fcw, const float* __restrict__ fcb,
    const float* __restrict__ dw,  const float* __restrict__ db,
    float* __restrict__ node_emb_o, float* __restrict__ recon_node_o,
    float* __restrict__ coord_out_o)
{
  __shared__ NodeScratch u;                 // 32 KB
  __shared__ __bf16 sW1[128 * 128];         // Wt[n][k], 32 KB
  __shared__ __bf16 sH [128 * 128];         // 32 KB
  __shared__ __bf16 sW2[64 * 128];          // 16 KB
  __shared__ float sEmb[128 * 8];
  __shared__ float sB1[128], sB2[64], sFCW[512], sFCB[8], sDW[512], sDB[64];

  const int t  = threadIdx.x;
  const int n0 = blockIdx.x * 128;

  copy_w(sW1, w1t, 128 * 128, t);
  copy_w(sW2, w2t, 64 * 128, t);
  if (t < 128) sB1[t] = b1[t];
  if (t < 64)  { sB2[t] = b2[t]; sDB[t] = db[t]; }
  if (t < 8)   sFCB[t] = fcb[t];
  for (int idx = t; idx < 512; idx += 256) { sFCW[idx] = fcw[idx]; sDW[idx] = dw[idx]; }

  const int il = t >> 1, hf = t & 1;
  const int i = n0 + il;
  const bool valid = i < NNODE;
  __bf16* Ai = u.A + il * 128;
  if (hf == 0) {
    #pragma unroll 8
    for (int k = 0; k < 64; ++k) Ai[k] = f2bf(valid ? node_feats[i * 64 + k] : 0.f);
  } else {
    #pragma unroll 8
    for (int k = 0; k < 32; ++k) Ai[64 + k] = f2bf(valid ? agg[i * 32 + k] : 0.f);
    #pragma unroll
    for (int j = 0; j < 3; ++j) {
      float c = valid ? coords[i * 3 + j] + coordacc[i * 3 + j] : 0.f;
      Ai[96 + j] = f2bf(c);
      if (valid) coord_out_o[i * 3 + j] = c;
    }
    #pragma unroll
    for (int k = 99; k < 128; ++k) Ai[k] = f2bf(0.f);
  }
  wait_async_copies();
  __syncthreads();

  const int wave = t >> 5, lane = t & 31;
  const int ln = lane & 15, gh = lane >> 4;
  const int m0 = wave * 16;

  for (int nt = 0; nt < 8; ++nt) {
    v8f acc; float bb = sB1[nt * 16 + ln];
    #pragma unroll
    for (int r = 0; r < 8; ++r) acc[r] = bb;
    #pragma unroll
    for (int kc = 0; kc < 4; ++kc) {
      v16bf a = lds_frag_A(u.A + m0 * 128 + kc * 32, 128);
      v16bf b = lds_frag_B(sW1 + (nt * 16) * 128 + kc * 32, 128);
      acc = wmma_bf16(a, b, acc);
    }
    #pragma unroll
    for (int r = 0; r < 8; ++r)
      sH[(m0 + gh * 8 + r) * 128 + nt * 16 + ln] = f2bf(fmaxf(acc[r], 0.f));
  }
  __syncthreads();

  for (int nt = 0; nt < 4; ++nt) {
    v8f acc; float bb = sB2[nt * 16 + ln];
    #pragma unroll
    for (int r = 0; r < 8; ++r) acc[r] = bb;
    #pragma unroll
    for (int kc = 0; kc < 4; ++kc) {
      v16bf a = lds_frag_A(sH + m0 * 128 + kc * 32, 128);
      v16bf b = lds_frag_B(sW2 + (nt * 16) * 128 + kc * 32, 128);
      acc = wmma_bf16(a, b, acc);
    }
    #pragma unroll
    for (int r = 0; r < 8; ++r)
      u.NO[(m0 + gh * 8 + r) * 64 + nt * 16 + ln] = acc[r];
  }
  __syncthreads();

  const float* no = u.NO + il * 64;
  if (hf == 0) {
    #pragma unroll
    for (int j = 0; j < 8; ++j) {
      float s = sFCB[j];
      #pragma unroll 8
      for (int k = 0; k < 64; ++k) s += no[k] * sFCW[k * 8 + j];
      sEmb[il * 8 + j] = s;
      if (valid) node_emb_o[(size_t)i * 8 + j] = s;
    }
  }
  __syncthreads();
  if (valid) {
    #pragma unroll
    for (int j = hf * 32; j < hf * 32 + 32; ++j) {
      float s = sDB[j];
      #pragma unroll
      for (int k = 0; k < 8; ++k) s += sEmb[il * 8 + k] * sDW[k * 64 + j];
      recon_node_o[(size_t)i * 64 + j] = s;
    }
  }
}

// ---------------------------------------------------------------------------
// Adjacency kernel: sigmoid(3*(r_i + r_j - 2*emb@emb^T) - 1), diag zeroed.
// 128x128 output block per workgroup; one WMMA per 16x16 tile (K padded 8->32)
// ---------------------------------------------------------------------------
__global__ __launch_bounds__(256) void adj_kernel(
    const float* __restrict__ emb, float* __restrict__ adj)
{
  __shared__ __bf16 sEA[128 * 32];
  __shared__ __bf16 sEB[128 * 32];
  __shared__ float sRA[128], sRB[128];

  const int t = threadIdx.x;
  const int r0 = blockIdx.y * 128, c0 = blockIdx.x * 128;

  if (t < 128) {
    int i = r0 + t; float rs = 0.f;
    #pragma unroll
    for (int k = 0; k < 8; ++k) {
      float v = (i < NNODE) ? emb[(size_t)i * 8 + k] : 0.f;
      rs += v * v; sEA[t * 32 + k] = f2bf(v);
    }
    #pragma unroll
    for (int k = 8; k < 32; ++k) sEA[t * 32 + k] = f2bf(0.f);
    sRA[t] = rs;
  } else {
    int tt = t - 128; int j = c0 + tt; float rs = 0.f;
    #pragma unroll
    for (int k = 0; k < 8; ++k) {
      float v = (j < NNODE) ? emb[(size_t)j * 8 + k] : 0.f;
      rs += v * v; sEB[tt * 32 + k] = f2bf(v);
    }
    #pragma unroll
    for (int k = 8; k < 32; ++k) sEB[tt * 32 + k] = f2bf(0.f);
    sRB[tt] = rs;
  }
  __syncthreads();

  const int wave = t >> 5, lane = t & 31;
  const int ln = lane & 15, gh = lane >> 4;
  const int mt = wave;
  v16bf a = lds_frag_A(sEA + mt * 16 * 32, 32);
  for (int nt = 0; nt < 8; ++nt) {
    v16bf b = lds_frag_B(sEB + nt * 16 * 32, 32);
    v8f acc;
    #pragma unroll
    for (int r = 0; r < 8; ++r) acc[r] = 0.f;
    acc = wmma_bf16(a, b, acc);
    #pragma unroll
    for (int r = 0; r < 8; ++r) {
      int mi = mt * 16 + gh * 8 + r, nj = nt * 16 + ln;
      int gi = r0 + mi, gj = c0 + nj;
      if (gi < NNODE && gj < NNODE) {
        float d2 = sRA[mi] + sRB[nj] - 2.f * acc[r];
        float x = 3.f * d2 - 1.f;
        float v = 1.f / (1.f + __expf(-x));
        if (gi == gj) v = 0.f;
        adj[(size_t)gi * NNODE + gj] = v;
      }
    }
  }
}

extern "C" void kernel_launch(void* const* d_in, const int* in_sizes, int n_in,
                              void* d_out, int out_size, void* d_ws, size_t ws_size,
                              hipStream_t stream) {
  const float* node_feats = (const float*)d_in[0];
  const int*   eidx       = (const int*)  d_in[1];
  const float* edge_attr  = (const float*)d_in[2];
  const float* coords     = (const float*)d_in[3];
  const float* edge_w1    = (const float*)d_in[4];
  const float* edge_b1    = (const float*)d_in[5];
  const float* edge_w2    = (const float*)d_in[6];
  const float* edge_b2    = (const float*)d_in[7];
  const float* coord_w1   = (const float*)d_in[8];
  const float* coord_b1   = (const float*)d_in[9];
  const float* coord_w2   = (const float*)d_in[10];
  const float* node_w1    = (const float*)d_in[11];
  const float* node_b1    = (const float*)d_in[12];
  const float* node_w2    = (const float*)d_in[13];
  const float* node_b2    = (const float*)d_in[14];
  const float* fc_node_w  = (const float*)d_in[15];
  const float* fc_node_b  = (const float*)d_in[16];
  const float* fc_edge_w  = (const float*)d_in[17];
  const float* fc_edge_b  = (const float*)d_in[18];
  const float* dec_node_w = (const float*)d_in[19];
  const float* dec_node_b = (const float*)d_in[20];
  const float* dec_edge_w = (const float*)d_in[21];
  const float* dec_edge_b = (const float*)d_in[22];

  float* ws        = (float*)d_ws;
  float* edge_out  = ws;                                   // E*32
  float* agg       = ws + (size_t)NEDGE * 32;              // N*32
  float* coordacc  = agg + (size_t)NNODE * 32;             // N*3
  __bf16* wbf      = (__bf16*)(coordacc + (size_t)NNODE * 3); // 53248 bf16
  __bf16* ew1t = wbf;
  __bf16* ew2t = wbf + 24576;
  __bf16* nw1t = wbf + 28672;
  __bf16* nw2t = wbf + 45056;

  float* out          = (float*)d_out;
  float* node_emb_o   = out;                               // N*8
  float* edge_emb_o   = node_emb_o + (size_t)NNODE * 8;    // E*8
  float* recon_node_o = edge_emb_o + (size_t)NEDGE * 8;    // N*64
  float* recon_edge_o = recon_node_o + (size_t)NNODE * 64; // E*32
  float* adj_o        = recon_edge_o + (size_t)NEDGE * 32; // N*N
  float* coord_out_o  = adj_o + (size_t)NNODE * NNODE;     // N*3

  int zn = NNODE * 32 + NNODE * 3;   // agg + coordacc are contiguous
  zero_kernel<<<(zn + 255) / 256, 256, 0, stream>>>(agg, zn);
  prep_weights<<<(53248 + 255) / 256, 256, 0, stream>>>(
      edge_w1, edge_w2, node_w1, node_w2, wbf);

  edge_kernel<<<NEDGE / 128, 256, 0, stream>>>(
      node_feats, eidx, edge_attr, coords,
      ew1t, edge_b1, ew2t, edge_b2,
      coord_w1, coord_b1, coord_w2,
      fc_edge_w, fc_edge_b, dec_edge_w, dec_edge_b,
      edge_out, agg, coordacc, edge_emb_o, recon_edge_o);

  node_kernel<<<(NNODE + 127) / 128, 256, 0, stream>>>(
      node_feats, coords, agg, coordacc,
      nw1t, node_b1, nw2t, node_b2,
      fc_node_w, fc_node_b, dec_node_w, dec_node_b,
      node_emb_o, recon_node_o, coord_out_o);

  dim3 agrid((NNODE + 127) / 128, (NNODE + 127) / 128);
  adj_kernel<<<agrid, 256, 0, stream>>>(node_emb_o, adj_o);
}
